// ADMEncoder_4063039062759
// MI455X (gfx1250) — matrix operand order repaired and checked by reading, not compiled, not run on hardware
//
#include <hip/hip_runtime.h>
#include <cstdint>
#include <cstddef>

// ---------------------------------------------------------------------------
// Types / constants
// ---------------------------------------------------------------------------
typedef __attribute__((ext_vector_type(16))) __bf16 v16bf;
typedef __attribute__((ext_vector_type(8)))  __bf16 v8bf;
typedef __attribute__((ext_vector_type(8)))  float  v8f;

constexpr int NRES  = 8192;
constexpr int KNB   = 32;
constexpr int HEADS = 8;

#define DEV __device__ __forceinline__

DEV float wredSum(float v){
#pragma unroll
  for (int o = 16; o > 0; o >>= 1) v += __shfl_xor(v, o, 32);
  return v;
}
DEV float wredMax(float v){
#pragma unroll
  for (int o = 16; o > 0; o >>= 1) v = fmaxf(v, __shfl_xor(v, o, 32));
  return v;
}
DEV float geluf(float x){
  const float c0 = 0.7978845608028654f;
  float t = tanhf(c0 * (x + 0.044715f * x * x * x));
  return 0.5f * x * (1.0f + t);
}

// ---------------------------------------------------------------------------
// Generic WMMA bf16 GEMM:  out[M,Nd] = act(A[M,Kd] @ W[Kd,Nd] + bias)
// Requirements (all call sites satisfy them): Kd % 32 == 0.
// A: bf16 row-major; W: f32 row-major.
// 64x64 tile / 128-thread block (4 waves; wave wv owns rows wv*16..wv*16+15).
// A fragments load straight from global (2x b128 per lane, fragment order ==
// contiguous K-octets).  W is staged to LDS transposed into fragment order,
// double-buffered so next-chunk staging overlaps current-chunk WMMAs; each
// lane reads its 16 B-values as one 32B (2x ds_load_b128) read.
// Padding columns (col >= Nd) hold clamped garbage; epilogue never stores them.
// ---------------------------------------------------------------------------
__global__ void __launch_bounds__(128)
k_gemm(const __bf16* __restrict__ A, const float* __restrict__ Wg,
       float* __restrict__ outF, __bf16* __restrict__ outB,
       const float* __restrict__ bias, int M, int Kd, int Nd, int act)
{
  __shared__ __align__(32) __bf16 Bfrag[2][64][40];  // [buf][col][k], 80B rows
  const int t    = threadIdx.x;
  const int lane = t & 31;
  const int wv   = t >> 5;
  const int lm   = lane & 15;
  const int hi   = lane >> 4;
  const int rowBase = blockIdx.x * 64;
  const int colBase = blockIdx.y * 64;

  int arow = rowBase + wv * 16 + lm;
  if (arow >= M) arow = M - 1;                 // all our M are tile multiples
  const __bf16* __restrict__ Arow = A + (size_t)arow * Kd;

  // stage one 32x64 W chunk (f32 -> bf16, transposed) into Bfrag[buf]
  auto stageW = [&](int k0, int buf){
#pragma unroll
    for (int it = 0; it < 8; it++){
      int item = it * 128 + t;                 // 1024 items: (rp 0..15, c 0..63)
      int rp = item >> 6, c = item & 63;
      int gk = k0 + 2 * rp;                    // Kd % 32 == 0 -> always in range
      int gc = colBase + c; if (gc >= Nd) gc = Nd - 1;   // clamp, no branch
      float f0 = Wg[(size_t)gk * Nd + gc];
      float f1 = Wg[(size_t)(gk + 1) * Nd + gc];
      union { __bf16 b[2]; unsigned u; } pk;
      pk.b[0] = (__bf16)f0; pk.b[1] = (__bf16)f1;
      *(unsigned*)&Bfrag[buf][c][2 * rp] = pk.u;         // one b32 LDS store
    }
  };

  v8f zz = {0.f,0.f,0.f,0.f,0.f,0.f,0.f,0.f};
  v8f acc[4];
#pragma unroll
  for (int i = 0; i < 4; i++) acc[i] = zz;

  stageW(0, 0);
  __syncthreads();

  int buf = 0;
  for (int k0 = 0; k0 < Kd; k0 += 32){
    if (k0 + 32 < Kd){
      stageW(k0 + 32, buf ^ 1);                // overlaps with WMMAs below
      __builtin_prefetch(Arow + k0 + 32, 0, 1);
    }
    // A fragment: two aligned 16B global loads in fragment order
    union { v16bf v; v8bf h[2]; } ua;
    ua.h[0] = *(const v8bf*)(Arow + k0 + hi * 16);       // K = hi*16 + [0..8)
    ua.h[1] = *(const v8bf*)(Arow + k0 + hi * 16 + 8);   // K = hi*16+8 + [0..8)

#pragma unroll
    for (int ns = 0; ns < 4; ns++){
      // B fragment: col = ns*16+lm, K = hi*16 + [0..16) contiguous in LDS
      v16bf bv = *(const v16bf*)&Bfrag[buf][ns * 16 + lm][hi * 16];
      acc[ns] = __builtin_amdgcn_wmma_f32_16x16x32_bf16(
          false, ua.v, false, bv, (short)0, acc[ns], false, false);
    }
    __syncthreads();                            // one barrier per chunk
    buf ^= 1;
  }

  // epilogue (C layout: vgpr g -> row g + hi*8, col = lane%16)
#pragma unroll
  for (int ns = 0; ns < 4; ns++){
    int col = colBase + ns * 16 + lm;
    if (col >= Nd) continue;
    float bvl = bias ? bias[col] : 0.f;
#pragma unroll
    for (int g = 0; g < 8; g++){
      int row = rowBase + wv * 16 + g + hi * 8;
      if (row >= M) continue;
      float v = acc[ns][g] + bvl;
      if (act == 1) v = geluf(v);
      size_t o = (size_t)row * Nd + col;
      if (outF) outF[o] = v;
      if (outB) outB[o] = (__bf16)v;
    }
  }
}

// ---------------------------------------------------------------------------
// Per-node geometry features -> F0[N,768] bf16 ( rd400 | rdir75 | lp15 |
// dssp3 | LN(prev)256 | pad19 ).  Also stores frames R and ca.
// One wave per node.
// ---------------------------------------------------------------------------
__global__ void k_features(const float* __restrict__ pos,
                           const float* __restrict__ prev, int n_total,
                           float* __restrict__ Rbuf, float* __restrict__ cabuf,
                           __bf16* __restrict__ F0,
                           const float* __restrict__ lps, const float* __restrict__ lpo)
{
  int wid  = (int)((blockIdx.x * blockDim.x + threadIdx.x) >> 5);
  int lane = threadIdx.x & 31;
  if (wid >= n_total) return;
  const float* pn = pos + (size_t)wid * 15;
  float nnx = pn[0],  nny = pn[1],  nnz = pn[2];
  float cax = pn[3],  cay = pn[4],  caz = pn[5];
  float ccx = pn[6],  ccy = pn[7],  ccz = pn[8];
  float e1x = ccx - cax, e1y = ccy - cay, e1z = ccz - caz;
  float n1 = sqrtf(e1x*e1x + e1y*e1y + e1z*e1z + 1e-12f);
  e1x /= n1; e1y /= n1; e1z /= n1;
  float v2x = nnx - cax, v2y = nny - cay, v2z = nnz - caz;
  float dd = e1x*v2x + e1y*v2y + e1z*v2z;
  float u2x = v2x - e1x*dd, u2y = v2y - e1y*dd, u2z = v2z - e1z*dd;
  float n2 = sqrtf(u2x*u2x + u2y*u2y + u2z*u2z + 1e-12f);
  float e2x = u2x/n2, e2y = u2y/n2, e2z = u2z/n2;
  float e3x = e1y*e2z - e1z*e2y;
  float e3y = e1z*e2x - e1x*e2z;
  float e3z = e1x*e2y - e1y*e2x;

  if (lane == 0){
    float* R = Rbuf + (size_t)wid * 9;      // R[d*3+e], column e = frame axis e
    R[0]=e1x; R[1]=e2x; R[2]=e3x;
    R[3]=e1y; R[4]=e2y; R[5]=e3y;
    R[6]=e1z; R[7]=e2z; R[8]=e3z;
    cabuf[wid*3+0]=cax; cabuf[wid*3+1]=cay; cabuf[wid*3+2]=caz;
  }

  auto lpos = [&](int a, float& x, float& y, float& z){
    float vx = pn[a*3+0]-cax, vy = pn[a*3+1]-cay, vz = pn[a*3+2]-caz;
    x = vx*e1x + vy*e1y + vz*e1z;
    y = vx*e2x + vy*e2y + vz*e2z;
    z = vx*e3x + vy*e3y + vz*e3z;
  };

  __bf16* Fr = F0 + (size_t)wid * 768;
  if (lane < 25){
    int a = lane / 5, b = lane % 5;
    float ax,ay,az,bx,by,bz;
    lpos(a,ax,ay,az); lpos(b,bx,by,bz);
    float rx = ax-bx, ry = ay-by, rz = az-bz;
    float d = sqrtf(rx*rx+ry*ry+rz*rz + 1e-12f);
    const float sig = 5.f/16.f;
#pragma unroll
    for (int j = 0; j < 16; j++){
      float c = 5.f * (float)j / 15.f;
      float q = (d - c) / sig;
      Fr[lane*16 + j] = (__bf16)expf(-q*q);
    }
    Fr[400 + lane*3 + 0] = (__bf16)(rx/d);
    Fr[400 + lane*3 + 1] = (__bf16)(ry/d);
    Fr[400 + lane*3 + 2] = (__bf16)(rz/d);
  }
  if (lane < 5){
    float x,y,z; lpos(lane,x,y,z);
    Fr[475 + lane*3 + 0] = (__bf16)(x/5.f);
    Fr[475 + lane*3 + 1] = (__bf16)(y/5.f);
    Fr[475 + lane*3 + 2] = (__bf16)(z/5.f);
  }
  if (lane == 0){
    float d3;
    if (wid + 3 < n_total){
      const float* pm = pos + (size_t)(wid+3) * 15;
      float dx = pm[3]-cax, dy = pm[4]-cay, dz = pm[5]-caz;
      d3 = sqrtf(dx*dx+dy*dy+dz*dz + 1e-12f);
    } else d3 = 20.f;
    int cls = (d3 < 7.f) ? 0 : ((d3 > 10.f) ? 1 : 2);
    for (int j = 0; j < 3; j++) Fr[490+j] = (__bf16)((j==cls)?1.f:0.f);
  }
  if (lane < 19) Fr[749 + lane] = (__bf16)0.f;

  // LN(prev) -> cols 493..748
  float xv[8]; float sum = 0.f;
#pragma unroll
  for (int e = 0; e < 8; e++){
    xv[e] = prev[(size_t)wid*256 + e*32 + lane];
    sum += xv[e];
  }
  float mean = wredSum(sum) / 256.f;
  float vs = 0.f;
#pragma unroll
  for (int e = 0; e < 8; e++){ float d = xv[e]-mean; vs += d*d; }
  float inv = rsqrtf(wredSum(vs)/256.f + 1e-5f);
#pragma unroll
  for (int e = 0; e < 8; e++){
    int c = e*32 + lane;
    Fr[493 + c] = (__bf16)((xv[e]-mean)*inv*lps[c] + lpo[c]);
  }
}

// ---------------------------------------------------------------------------
// Per-edge pair features -> PF[N*K,128] bf16 (onehot65 | dist16 | dir15 |
// rot9 | pad23) and pair_mask.  One thread per edge.
// ---------------------------------------------------------------------------
__global__ void k_pairfeat(const float* __restrict__ pos,
                           const float* __restrict__ Rbuf,
                           const float* __restrict__ cabuf,
                           const int* __restrict__ nbrs,
                           const int* __restrict__ resi,
                           const int* __restrict__ chain,
                           const unsigned char* __restrict__ mask, int n_total,
                           __bf16* __restrict__ PF, float* __restrict__ pmask)
{
  long long e = (long long)blockIdx.x * blockDim.x + threadIdx.x;
  if (e >= (long long)n_total * KNB) return;
  int n = (int)(e >> 5);
  int nb = nbrs[e];
  int safe = nb > 0 ? nb : 0;
  float valid = (nb != -1) ? 1.f : 0.f;
  float pm = (mask[n] ? 1.f : 0.f) * (mask[safe] ? 1.f : 0.f) * valid;
  pmask[e] = pm;

  __bf16* dst = PF + (size_t)e * 128;
  for (int i = 0; i < 128; i++) dst[i] = (__bf16)0.f;

  int rs = resi[safe] - resi[n];
  rs = (rs < -32) ? -32 : (rs > 32 ? 32 : rs);
  rs += 32;
  float same = ((chain[safe] == chain[n]) && (nb != -1)) ? 1.f : 0.f;
  dst[rs] = (__bf16)same;

  float c0x = cabuf[n*3+0], c0y = cabuf[n*3+1], c0z = cabuf[n*3+2];
  float dx = cabuf[safe*3+0]-c0x, dy = cabuf[safe*3+1]-c0y, dz = cabuf[safe*3+2]-c0z;
  float d = sqrtf(dx*dx+dy*dy+dz*dz + 1e-12f);
  const float sig = 22.f/16.f;
#pragma unroll
  for (int j = 0; j < 16; j++){
    float c = 22.f*(float)j/15.f;
    float q = (d-c)/sig;
    dst[65+j] = (__bf16)expf(-q*q);
  }
  const float* Rn = Rbuf + (size_t)n * 9;
  const float* Rm = Rbuf + (size_t)safe * 9;
  const float* pm5 = pos + (size_t)safe * 15;
#pragma unroll
  for (int a = 0; a < 5; a++){
    float vx = pm5[a*3+0]-c0x, vy = pm5[a*3+1]-c0y, vz = pm5[a*3+2]-c0z;
    float lx = vx*Rn[0] + vy*Rn[3] + vz*Rn[6];
    float ly = vx*Rn[1] + vy*Rn[4] + vz*Rn[7];
    float lz = vx*Rn[2] + vy*Rn[5] + vz*Rn[8];
    float nr = sqrtf(lx*lx+ly*ly+lz*lz + 1e-12f);
    dst[81+a*3+0] = (__bf16)(lx/nr);
    dst[81+a*3+1] = (__bf16)(ly/nr);
    dst[81+a*3+2] = (__bf16)(lz/nr);
  }
#pragma unroll
  for (int a = 0; a < 3; a++)
#pragma unroll
    for (int b = 0; b < 3; b++){
      float s = Rn[0*3+a]*Rm[0*3+b] + Rn[1*3+a]*Rm[1*3+b] + Rn[2*3+a]*Rm[2*3+b];
      dst[96+a*3+b] = (__bf16)s;
    }
}

// nb_mean[N,16]  (nb_dist_f == dist_f by orthonormality of R). Wave per node.
__global__ void k_nbmean(const float* __restrict__ cabuf,
                         const int* __restrict__ nbrs,
                         const float* __restrict__ pmask, int n_total,
                         float* __restrict__ nbmean)
{
  int wid  = (int)((blockIdx.x * blockDim.x + threadIdx.x) >> 5);
  int lane = threadIdx.x & 31;
  if (wid >= n_total) return;
  long long e = (long long)wid * KNB + lane;
  int nb = nbrs[e];
  int safe = nb > 0 ? nb : 0;
  float pm = pmask[e];
  float dx = cabuf[safe*3+0]-cabuf[wid*3+0];
  float dy = cabuf[safe*3+1]-cabuf[wid*3+1];
  float dz = cabuf[safe*3+2]-cabuf[wid*3+2];
  float d = sqrtf(dx*dx+dy*dy+dz*dz + 1e-12f);
  float cnt = fmaxf(wredSum(pm), 1.f);
  const float sig = 22.f/16.f;
#pragma unroll
  for (int j = 0; j < 16; j++){
    float c = 22.f*(float)j/15.f;
    float q = (d-c)/sig;
    float s = wredSum(expf(-q*q)*pm);
    if (lane == 0) nbmean[wid*16+j] = s/cnt;
  }
}

// LayerNorm rows (wave per row).  Input f32 or bf16; up to two f32 outs + bf16.
__global__ void k_ln(const float* __restrict__ inF, const __bf16* __restrict__ inB,
                     float* __restrict__ outF, float* __restrict__ outF2,
                     __bf16* __restrict__ outB,
                     const float* __restrict__ s, const float* __restrict__ o,
                     int C, long long M)
{
  long long wid = (long long)(blockIdx.x * blockDim.x + threadIdx.x) >> 5;
  int lane = threadIdx.x & 31;
  if (wid >= M) return;
  int per = C >> 5;
  float xv[8]; float sum = 0.f;
  for (int e = 0; e < per; e++){
    size_t idx = (size_t)wid * C + e*32 + lane;
    float v = inF ? inF[idx] : (float)inB[idx];
    xv[e] = v; sum += v;
  }
  float mean = wredSum(sum) / (float)C;
  float vs = 0.f;
  for (int e = 0; e < per; e++){ float d = xv[e]-mean; vs += d*d; }
  float inv = rsqrtf(wredSum(vs)/(float)C + 1e-5f);
  for (int e = 0; e < per; e++){
    int c = e*32 + lane;
    size_t idx = (size_t)wid * C + c;
    float y = (xv[e]-mean)*inv*s[c] + o[c];
    if (outF)  outF[idx]  = y;
    if (outF2) outF2[idx] = y;
    if (outB)  outB[idx]  = (__bf16)y;
  }
}

// residual + LN over D=256:  mode 1: u = updA*maskf ; mode 2: u = updA + updB*maskf.
// local = LN(local+u); inc += u; localbf = bf16(local).  Wave per row.
__global__ void k_resid_ln(float* __restrict__ local, const float* __restrict__ updA,
                           const float* __restrict__ updB,
                           const unsigned char* __restrict__ mask, int mode,
                           float* __restrict__ inc,
                           const float* __restrict__ s, const float* __restrict__ o,
                           __bf16* __restrict__ localbf, int M)
{
  int wid  = (int)((blockIdx.x * blockDim.x + threadIdx.x) >> 5);
  int lane = threadIdx.x & 31;
  if (wid >= M) return;
  float mf = mask[wid] ? 1.f : 0.f;
  float xv[8]; float sum = 0.f;
#pragma unroll
  for (int e = 0; e < 8; e++){
    size_t idx = (size_t)wid * 256 + e*32 + lane;
    float u = (mode == 1) ? updA[idx]*mf : (updA[idx] + updB[idx]*mf);
    float li = local[idx] + u;
    inc[idx] += u;
    xv[e] = li; sum += li;
  }
  float mean = wredSum(sum) / 256.f;
  float vs = 0.f;
#pragma unroll
  for (int e = 0; e < 8; e++){ float d = xv[e]-mean; vs += d*d; }
  float inv = rsqrtf(wredSum(vs)/256.f + 1e-5f);
#pragma unroll
  for (int e = 0; e < 8; e++){
    int c = e*32 + lane;
    size_t idx = (size_t)wid * 256 + c;
    float y = (xv[e]-mean)*inv*s[c] + o[c];
    local[idx] = y;
    localbf[idx] = (__bf16)y;
  }
}

// out = local + LN(inc).  Wave per row.
__global__ void k_final(const float* __restrict__ local, const float* __restrict__ inc,
                        const float* __restrict__ s, const float* __restrict__ o,
                        float* __restrict__ out, int M)
{
  int wid  = (int)((blockIdx.x * blockDim.x + threadIdx.x) >> 5);
  int lane = threadIdx.x & 31;
  if (wid >= M) return;
  float xv[8]; float sum = 0.f;
#pragma unroll
  for (int e = 0; e < 8; e++){
    xv[e] = inc[(size_t)wid*256 + e*32 + lane]; sum += xv[e];
  }
  float mean = wredSum(sum) / 256.f;
  float vs = 0.f;
#pragma unroll
  for (int e = 0; e < 8; e++){ float d = xv[e]-mean; vs += d*d; }
  float inv = rsqrtf(wredSum(vs)/256.f + 1e-5f);
#pragma unroll
  for (int e = 0; e < 8; e++){
    int c = e*32 + lane;
    size_t idx = (size_t)wid*256 + c;
    out[idx] = local[idx] + (xv[e]-mean)*inv*s[c] + o[c];
  }
}

// Attention: wave per (n,h); lane == neighbour slot k (K==32==wave32).
__global__ void k_attn(const float* __restrict__ qf, const __bf16* __restrict__ kbf,
                       const __bf16* __restrict__ vbf, const __bf16* __restrict__ pvbf,
                       const float* __restrict__ pbias, const float* __restrict__ pmask,
                       const int* __restrict__ nbrs, int n_total,
                       __bf16* __restrict__ obf)
{
  int wid  = (int)((blockIdx.x * blockDim.x + threadIdx.x) >> 5);
  int lane = threadIdx.x & 31;
  if (wid >= n_total * HEADS) return;
  int n = wid >> 3, h = wid & 7;
  long long e = (long long)n * KNB + lane;
  int nb = nbrs[e];
  int safe = nb > 0 ? nb : 0;
  float pm = pmask[e];
  float dot = 0.f;
#pragma unroll
  for (int d = 0; d < 16; d++){
    float qv = qf[(size_t)n*128 + h*16 + d];
    float kv = (float)kbf[(size_t)safe*128 + h*16 + d];
    dot += qv * kv;
  }
  float logit = dot * 0.25f + pbias[(size_t)e*8 + h];
  if (!(pm > 0.f)) logit = -1e9f;
  float mx = wredMax(logit);
  float ex = expf(logit - mx);
  float sm = wredSum(ex);
  float w = ex / sm;
#pragma unroll
  for (int d = 0; d < 16; d++){
    float val = w * ((float)vbf[(size_t)safe*128 + h*16 + d]
                   + (float)pvbf[(size_t)e*128 + h*16 + d]);
    float r = wredSum(val);
    if (lane == 0) obf[(size_t)n*128 + h*16 + d] = (__bf16)r;
  }
}

// small utility kernels -----------------------------------------------------
__global__ void k_zero(float* __restrict__ p, long long n){
  long long i = (long long)blockIdx.x * blockDim.x + threadIdx.x;
  if (i < n) p[i] = 0.f;
}
__global__ void k_copyw(float* __restrict__ dst, const float* __restrict__ src,
                        int rows, int cols, int rowoff){
  long long i = (long long)blockIdx.x * blockDim.x + threadIdx.x;
  if (i >= (long long)rows * cols) return;
  int r = (int)(i / cols), c = (int)(i % cols);
  dst[(size_t)(rowoff + r) * cols + c] = src[i];
}
__global__ void k_concat(const __bf16* __restrict__ localbf,
                         const float* __restrict__ nbmean, int n_total,
                         __bf16* __restrict__ Cc){
  long long i = (long long)blockIdx.x * blockDim.x + threadIdx.x;
  if (i >= (long long)n_total * 288) return;
  int n = (int)(i / 288), c = (int)(i % 288);
  __bf16 v;
  if (c < 256)      v = localbf[(size_t)n*256 + c];
  else if (c < 272) v = (__bf16)nbmean[n*16 + (c-256)];
  else              v = (__bf16)0.f;
  Cc[i] = v;
}
__global__ void k_counts(const int* __restrict__ batch, const int* __restrict__ chain,
                         const unsigned char* __restrict__ mask, int n_total,
                         float* __restrict__ bcnt, float* __restrict__ ccnt){
  int n = blockIdx.x * blockDim.x + threadIdx.x;
  if (n >= n_total) return;
  float mf = mask[n] ? 1.f : 0.f;
  atomicAdd(&bcnt[batch[n]], mf);
  atomicAdd(&ccnt[chain[n]], mf);
}
__global__ void k_segacc(const float* __restrict__ lu, const int* __restrict__ batch,
                         const int* __restrict__ chain,
                         const unsigned char* __restrict__ mask, int n_total,
                         float* __restrict__ bsum, float* __restrict__ csum){
  long long i = (long long)blockIdx.x * blockDim.x + threadIdx.x;
  if (i >= (long long)n_total * 512) return;
  int n = (int)(i >> 9), c = (int)(i & 511);
  float v = lu[i] * (mask[n] ? 1.f : 0.f);
  atomicAdd(&bsum[(size_t)batch[n]*512 + c], v);
  atomicAdd(&csum[(size_t)chain[n]*512 + c], v);
}
__global__ void k_gate(const float* __restrict__ gb, const float* __restrict__ gc,
                       const float* __restrict__ gl, const float* __restrict__ lu,
                       const int* __restrict__ batch, const int* __restrict__ chain,
                       const float* __restrict__ bsum, const float* __restrict__ bcnt,
                       const float* __restrict__ csum, const float* __restrict__ ccnt,
                       int n_total, __bf16* __restrict__ hidden){
  long long i = (long long)blockIdx.x * blockDim.x + threadIdx.x;
  if (i >= (long long)n_total * 512) return;
  int n = (int)(i >> 9), c = (int)(i & 511);
  float bm = bsum[(size_t)batch[n]*512 + c] / fmaxf(bcnt[batch[n]], 1.f);
  float cm = csum[(size_t)chain[n]*512 + c] / fmaxf(ccnt[chain[n]], 1.f);
  hidden[i] = (__bf16)(gb[i]*bm + gc[i]*cm + gl[i]*lu[i]);
}

// ---------------------------------------------------------------------------
// Host launcher
// ---------------------------------------------------------------------------
extern "C" void kernel_launch(void* const* d_in, const int* in_sizes, int n_in,
                              void* d_out, int out_size, void* d_ws, size_t ws_size,
                              hipStream_t stream)
{
  (void)in_sizes; (void)n_in; (void)out_size; (void)ws_size;
  const int N = NRES;
  const long long NK = (long long)N * KNB;

  const float*         pos   = (const float*)d_in[0];
  const int*           nbrs  = (const int*)d_in[1];
  const int*           resi  = (const int*)d_in[2];
  const int*           chain = (const int*)d_in[3];
  const int*           batch = (const int*)d_in[4];
  const unsigned char* mask  = (const unsigned char*)d_in[5];
  const float*         prev  = (const float*)d_in[6];
  auto Pf = [&](int i){ return (const float*)d_in[i]; };

  // ---- workspace layout ----
  char* wsc = (char*)d_ws;
  size_t off = 0;
  auto alloc = [&](size_t bytes){ void* p = wsc + off; off = (off + bytes + 255) & ~(size_t)255; return p; };
  auto AF = [&](size_t n){ return (float*) alloc(n * sizeof(float)); };
  auto AB = [&](size_t n){ return (__bf16*)alloc(n * sizeof(__bf16)); };

  float*  Rbuf   = AF((size_t)N*9);
  float*  cabuf  = AF((size_t)N*3);
  __bf16* F0     = AB((size_t)N*768);
  float*  Wcat   = AF(768*256);
  float*  Wp     = AF(128*64);
  float*  Wt2    = AF(288*512);
  float*  t0     = AF((size_t)N*256);
  float*  localF = AF((size_t)N*256);
  float*  incF   = AF((size_t)N*256);
  __bf16* localB = AB((size_t)N*256);
  __bf16* xbf    = AB((size_t)N*256);
  __bf16* h1bf   = AB((size_t)N*1024);
  __bf16* PF     = AB((size_t)NK*128);
  __bf16* PB64   = AB((size_t)NK*64);
  __bf16* PB128  = AB((size_t)NK*128);
  float*  pbias  = AF((size_t)NK*8);
  float*  pmask  = AF((size_t)NK);
  float*  nbmean = AF((size_t)N*16);
  float*  qf     = AF((size_t)N*128);
  __bf16* kbf    = AB((size_t)N*128);
  __bf16* vbf    = AB((size_t)N*128);
  __bf16* obf    = AB((size_t)N*128);
  float*  updA   = AF((size_t)N*256);
  __bf16* Cc     = AB((size_t)N*288);
  __bf16* thbf   = AB((size_t)N*512);
  float*  tpf    = AF((size_t)N*256);
  float*  luf    = AF((size_t)N*512);
  float*  gb_b   = AF((size_t)N*512);
  float*  gb_c   = AF((size_t)N*512);
  float*  gb_l   = AF((size_t)N*512);
  __bf16* hidbf  = AB((size_t)N*512);
  float*  segs   = AF(8*512 + 16*512 + 8 + 16);
  float*  bsum = segs;
  float*  csum = segs + 8*512;
  float*  bcnt = segs + 8*512 + 16*512;
  float*  ccnt = bcnt + 8;

  auto gE = [](long long n, int b){ return dim3((unsigned)((n + b - 1) / b)); };
  auto gemm = [&](const __bf16* A, const float* Wg, float* oF, __bf16* oB,
                  const float* bias, int M, int Kd, int Nd, int act){
    dim3 g((unsigned)((M + 63) / 64), (unsigned)((Nd + 63) / 64));
    k_gemm<<<g, 128, 0, stream>>>(A, Wg, oF, oB, bias, M, Kd, Nd, act);
  };
  auto zero = [&](float* p, long long n){ k_zero<<<gE(n,256),256,0,stream>>>(p, n); };
  auto copyw = [&](float* dst, const float* src, int r, int c, int ro){
    k_copyw<<<gE((long long)r*c,256),256,0,stream>>>(dst, src, r, c, ro);
  };

  // ---- prepare: features ----
  k_features<<<gE((long long)N*32,256),256,0,stream>>>(pos, prev, N, Rbuf, cabuf, F0, Pf(12), Pf(13));
  k_pairfeat<<<gE(NK,256),256,0,stream>>>(pos, Rbuf, cabuf, nbrs, resi, chain, mask, N, PF, pmask);
  k_nbmean  <<<gE((long long)N*32,256),256,0,stream>>>(cabuf, nbrs, pmask, N, nbmean);

  // prepare: local = LN1( gelu(LN0(F0@Wcat) @ mw1) @ mw2 ); inc = local
  zero(Wcat, 768*256);
  copyw(Wcat, Pf(7),  400, 256, 0);
  copyw(Wcat, Pf(8),   75, 256, 400);
  copyw(Wcat, Pf(9),   15, 256, 475);
  copyw(Wcat, Pf(10),   3, 256, 490);
  copyw(Wcat, Pf(11), 256, 256, 493);
  gemm(F0, Wcat, t0, nullptr, nullptr, N, 768, 256, 0);
  k_ln<<<gE((long long)N*32,256),256,0,stream>>>(t0, nullptr, nullptr, nullptr, xbf, Pf(14), Pf(15), 256, N);
  gemm(xbf, Pf(16), nullptr, h1bf, nullptr, N, 256, 1024, 1);
  gemm(h1bf, Pf(17), t0, nullptr, nullptr, N, 1024, 256, 0);
  k_ln<<<gE((long long)N*32,256),256,0,stream>>>(t0, nullptr, localF, incF, localB, Pf(18), Pf(19), 256, N);

  // ---- blocks ----
  for (int b = 0; b < 2; b++){
    int B = 20 + b * 28;
    // pair stack
    zero(Wp, 128*64);
    copyw(Wp, Pf(B+0), 65, 64, 0);
    copyw(Wp, Pf(B+1), 16, 64, 65);
    copyw(Wp, Pf(B+2), 15, 64, 81);
    copyw(Wp, Pf(B+3),  9, 64, 96);
    gemm(PF, Wp, nullptr, PB64, nullptr, (int)NK, 128, 64, 0);
    k_ln<<<gE(NK*32,256),256,0,stream>>>(nullptr, PB64, nullptr, nullptr, PB64, Pf(B+4), Pf(B+5), 64, NK);
    gemm(PB64,  Pf(B+6), nullptr, PB128, Pf(B+7), (int)NK, 64, 128, 1);
    gemm(PB128, Pf(B+8), nullptr, PB64,  Pf(B+9), (int)NK, 128, 64, 0);
    gemm(PB64,  Pf(B+13), pbias,  nullptr, nullptr, (int)NK, 64, 8,   0);
    gemm(PB64,  Pf(B+14), nullptr, PB128,  nullptr, (int)NK, 64, 128, 0); // pv

    // attention
    gemm(localB, Pf(B+10), qf, nullptr, nullptr, N, 256, 128, 0);
    gemm(localB, Pf(B+11), nullptr, kbf, nullptr, N, 256, 128, 0);
    gemm(localB, Pf(B+12), nullptr, vbf, nullptr, N, 256, 128, 0);
    k_attn<<<gE((long long)N*HEADS*32,256),256,0,stream>>>(qf, kbf, vbf, PB128, pbias, pmask, nbrs, N, obf);
    gemm(obf, Pf(B+15), updA, nullptr, nullptr, N, 128, 256, 0);
    k_resid_ln<<<gE((long long)N*32,256),256,0,stream>>>(localF, updA, nullptr, mask, 1, incF,
                                                         Pf(B+16), Pf(B+17), localB, N);

    // transition / gated update
    k_concat<<<gE((long long)N*288,256),256,0,stream>>>(localB, nbmean, N, Cc);
    zero(Wt2, 288*512);
    copyw(Wt2, Pf(B+18), 272, 512, 0);
    gemm(Cc, Wt2, nullptr, thbf, nullptr, N, 288, 512, 1);
    gemm(thbf, Pf(B+19), tpf, nullptr, nullptr, N, 512, 256, 0);
    gemm(localB, Pf(B+20), luf,  nullptr, nullptr, N, 256, 512, 0);
    gemm(localB, Pf(B+23), gb_b, nullptr, nullptr, N, 256, 512, 1); // w_bg
    gemm(localB, Pf(B+22), gb_c, nullptr, nullptr, N, 256, 512, 1); // w_cg
    gemm(localB, Pf(B+21), gb_l, nullptr, nullptr, N, 256, 512, 1); // w_lg
    zero(segs, 8*512 + 16*512 + 8 + 16);
    k_counts<<<gE(N,256),256,0,stream>>>(batch, chain, mask, N, bcnt, ccnt);
    k_segacc<<<gE((long long)N*512,256),256,0,stream>>>(luf, batch, chain, mask, N, bsum, csum);
    k_gate<<<gE((long long)N*512,256),256,0,stream>>>(gb_b, gb_c, gb_l, luf, batch, chain,
                                                      bsum, bcnt, csum, ccnt, N, hidbf);
    gemm(hidbf, Pf(B+24), updA, nullptr, Pf(B+25), N, 512, 256, 0);
    k_resid_ln<<<gE((long long)N*32,256),256,0,stream>>>(localF, updA, tpf, mask, 2, incF,
                                                         Pf(B+26), Pf(B+27), localB, N);
  }

  // ---- output ----
  k_final<<<gE((long long)N*32,256),256,0,stream>>>(localF, incF, Pf(76), Pf(77), (float*)d_out, N);
}
